// MpnnIntNet_9655086481676
// MI455X (gfx1250) — matrix operand order repaired and checked by reading, not compiled, not run on hardware
//
#include <hip/hip_runtime.h>

// MPNN (intnet) forward for MI455X / gfx1250, wave32 + v_wmma_f32_16x16x32_f16
// + GLOBAL_LOAD_ASYNC_TO_LDS_B128 (ASYNCcnt) for all big LDS weight fills.
// B=8, N=128, D=128, DE=32, OM=OU=128, T=12, L=3. Hidden dims (128,256,128).
//
// Pipeline per layer l:
//   k_pvpw    : Pvb = h@W0[:,0:128]+b0 (f32), Pw = h@W0[:,128:256] (f16)   [1024x128]
//   k_message : fused edge-MLP chain + gated sum over w  -> m_agg [1024x128] f32
//   k_update  : fused update MLP cat(h,m) 256->128->256->128->128 -> h' f16
// Readout: sum over nodes + 128->128->256->128->12 MLP (single WG).
// Weights are transposed+converted to f16 in workspace once per launch.
// Param flattening assumed: dict insertion order (message[0..2], update[0..2],
// readout), each MLP = (W0,b0,W1,b1,W2,b2,W3,b3).

typedef __attribute__((ext_vector_type(16))) _Float16 v16h;
typedef __attribute__((ext_vector_type(8)))  _Float16 v8h;
typedef __attribute__((ext_vector_type(8)))  float    v8f;
typedef __attribute__((ext_vector_type(4)))  float    v4f;

#define DEVINL static __device__ __forceinline__

// ---- CDNA5 async global->LDS copy (VGLOBAL GVS form, tracked by ASYNCcnt) ----
// 16 bytes per lane, no VGPR round trip:  LDS[lds_addr] = MEM[base + byte_off]
DEVINL void async_ld16(unsigned int lds_addr, unsigned int byte_off, const void* base) {
  asm volatile("global_load_async_to_lds_b128 %0, %1, %2"
               :: "v"(lds_addr), "v"(byte_off), "s"(base)
               : "memory");
}
DEVINL void async_wait() {
  asm volatile("s_wait_asynccnt 0x0" ::: "memory");
}
// generic->LDS address: low 32 bits of the flat address are the LDS offset
DEVINL unsigned int lds_addr_of(const void* p) {
  return (unsigned int)(unsigned long long)p;
}
// contiguous f16 block: dst_lds[i] = src[i]
DEVINL void copy_async_f16(_Float16* dst, const _Float16* src, int nelem,
                           int tid, int nthr) {
  const unsigned int dbase = lds_addr_of(dst);
  const int nv = nelem >> 3;
  for (int i = tid; i < nv; i += nthr)
    async_ld16(dbase + (unsigned int)i * 16u, (unsigned int)i * 16u, src);
}

// A fragment: 16x32 f16 tile at (m0,k0) of row-major [*][ld].
// lane<16: K = k0+[0..7], k0+[16..23]; lane>=16: K = k0+[8..15], k0+[24..31].
DEVINL v16h frag_a(const _Float16* p, int ld, int m0, int k0, int lane) {
  const int half = lane >> 4;
  const _Float16* base = p + (m0 + (lane & 15)) * ld + k0 + half * 8;
  v8h lo = *(const v8h*)(base);
  v8h hi = *(const v8h*)(base + 16);
  v16h r;
#pragma unroll
  for (int i = 0; i < 8; ++i) { r[i] = lo[i]; r[i + 8] = hi[i]; }
  return r;
}

// B fragment: 32x16 tile (K x N) at (k0, n0), from transposed weights Wt[N][K].
// lane<16: col n0+lane, K = k0+[0..7], k0+[16..23]; lane>=16: col n0+lane-16, K = k0+[8..15], k0+[24..31].
DEVINL v16h frag_b(const _Float16* wt, int ld, int n0, int k0, int lane) {
  const _Float16* base = wt + (n0 + (lane & 15)) * ld + k0 + (lane >> 4) * 8;
  v8h lo = *(const v8h*)(base);
  v8h hi = *(const v8h*)(base + 16);
  v16h r;
#pragma unroll
  for (int i = 0; i < 8; ++i) { r[i] = lo[i]; r[i + 8] = hi[i]; }
  return r;
}

DEVINL v8f wmma_f16(v16h a, v16h b, v8f c) {
  return __builtin_amdgcn_wmma_f32_16x16x32_f16(false, a, false, b, (short)0, c,
                                                false, false);
}

// D layout: VGPR r -> m = m0 + r + 8*(lane>=16), n = n0 + (lane&15)
DEVINL void store_relu(_Float16* x, int ld, int m0, int n0, int lane, v8f acc) {
  const int half = lane >> 4, ln = lane & 15;
#pragma unroll
  for (int r = 0; r < 8; ++r) {
    float v = acc[r];
    v = v > 0.f ? v : 0.f;
    x[(m0 + r + 8 * half) * ld + n0 + ln] = (_Float16)v;
  }
}

// ---------------- prep kernels ----------------

__global__ __launch_bounds__(256) void k_cvt_f16(const float* __restrict__ s,
                                                 _Float16* __restrict__ d, int n) {
  for (int i = blockIdx.x * blockDim.x + threadIdx.x; i < n;
       i += gridDim.x * blockDim.x)
    d[i] = (_Float16)s[i];
}

// W [K][N] f32 -> Wt [Npad][K] f16 (zero-padded rows N..Npad-1)
__global__ __launch_bounds__(256) void k_prep_wt(const float* __restrict__ W,
                                                 _Float16* __restrict__ Wt,
                                                 int K, int N, int Npad) {
  const int total = Npad * K;
  for (int i = blockIdx.x * blockDim.x + threadIdx.x; i < total;
       i += gridDim.x * blockDim.x) {
    const int n = i / K, k = i - n * K;
    Wt[i] = (_Float16)((n < N) ? W[k * N + n] : 0.f);
  }
}

// ---------------- Pv / Pw projection: h[1024][128] @ W0t ----------------
// grid 8 x 256 threads (8 waves x 16-row strips)

__global__ __launch_bounds__(256) void k_pvpw(const _Float16* __restrict__ h,
                                              const _Float16* __restrict__ W0t, // [128][288]
                                              const float* __restrict__ b0,
                                              float* __restrict__ Pvb,
                                              _Float16* __restrict__ Pw) {
  extern __shared__ char smem[];
  _Float16* w = (_Float16*)smem; // [128][256]  (K cols 0..255 of W0t)
  const int tid = threadIdx.x, lane = tid & 31, wid = tid >> 5;
  const int half = lane >> 4, ln = lane & 15;
  {
    const unsigned int wb = lds_addr_of(w);
    for (int i = tid; i < 128 * 32; i += 256) {
      const int n = i >> 5, c = i & 31;
      async_ld16(wb + (unsigned int)(n * 256 + c * 8) * 2u,
                 (unsigned int)(n * 288 + c * 8) * 2u, W0t);
    }
  }
  async_wait();
  __syncthreads();
  const int m0 = blockIdx.x * 128 + wid * 16;
  v16h a[4];
#pragma unroll
  for (int k = 0; k < 4; ++k) a[k] = frag_a(h, 128, m0, k * 32, lane);
  for (int nt = 0; nt < 8; ++nt) {
    const int n0 = nt * 16;
    const float bias = b0[n0 + ln];
    v8f av, aw;
#pragma unroll
    for (int r = 0; r < 8; ++r) { av[r] = bias; aw[r] = 0.f; }
#pragma unroll
    for (int k = 0; k < 4; ++k) {
      av = wmma_f16(a[k], frag_b(w, 256, n0, k * 32, lane), av);
      aw = wmma_f16(a[k], frag_b(w, 256, n0, 128 + k * 32, lane), aw);
    }
#pragma unroll
    for (int r = 0; r < 8; ++r) {
      const int m = m0 + r + 8 * half;
      Pvb[m * 128 + n0 + ln] = av[r];
      Pw[m * 128 + n0 + ln] = (_Float16)aw[r];
    }
  }
}

// ---------------- fused message MLP + gated reduction ----------------
// grid 256 (b = blk>>5 fixed per WG, 4 v's per WG), 256 threads (8 waves).
// All message weights + Pw strip resident in LDS (~300KB of the 320KB WGP LDS),
// filled with global_load_async_to_lds_b128 (ASYNCcnt) instead of VGPR bounces.

__global__ __launch_bounds__(256) void k_message(
    const float* __restrict__ e, const float* __restrict__ g,
    const float* __restrict__ Pvb, const _Float16* __restrict__ Pw,
    const _Float16* __restrict__ W0t,  // [128][288], cols 256..287 = edge part
    const _Float16* __restrict__ W1t,  // [256][128]
    const _Float16* __restrict__ W2t,  // [128][256]
    const _Float16* __restrict__ W3t,  // [128][128]
    const float* __restrict__ b1, const float* __restrict__ b2,
    const float* __restrict__ b3, float* __restrict__ magg) {
  extern __shared__ char smem[];
  _Float16* w1 = (_Float16*)smem;      // 256*128
  _Float16* w2 = w1 + 256 * 128;       // 128*256
  _Float16* w3 = w2 + 128 * 256;       // 128*128
  _Float16* wc = w3 + 128 * 128;       // 128*32
  _Float16* pw = wc + 128 * 32;        // 128*128
  _Float16* x1 = pw + 128 * 128;       // 128*128 (also x3)
  _Float16* x2 = x1 + 128 * 128;       // 128*256
  float* macc = (float*)(x2 + 128 * 256); // 128

  const int tid = threadIdx.x, lane = tid & 31, wid = tid >> 5;
  const int half = lane >> 4, ln = lane & 15;
  const int b = blockIdx.x >> 5, vg = blockIdx.x & 31;
  const int m0 = wid * 16; // w-strip for this wave

  copy_async_f16(w1, W1t, 256 * 128, tid, 256);
  copy_async_f16(w2, W2t, 128 * 256, tid, 256);
  copy_async_f16(w3, W3t, 128 * 128, tid, 256);
  {
    const unsigned int wcb = lds_addr_of(wc);
    for (int i = tid; i < 128 * 4; i += 256) { // edge slice of W0t (stride 288)
      const int n = i >> 2, c = i & 3;
      async_ld16(wcb + (unsigned int)(n * 32 + c * 8) * 2u,
                 (unsigned int)(n * 288 + 256 + c * 8) * 2u, W0t);
    }
  }
  copy_async_f16(pw, Pw + b * 128 * 128, 128 * 128, tid, 256);
  async_wait();

  for (int vi = 0; vi < 4; ++vi) {
    const int v = vg * 4 + vi;
    const int bv = b * 128 + v;
    if (vi < 3) { // prefetch next v's 16KB e-block into cache (global_prefetch)
      const float* pf = e + ((size_t)(bv + 1) * 128) * 32 + tid * 16;
      __builtin_prefetch(pf, 0, 3);
    }
    __syncthreads();
    if (tid < 128) macc[tid] = 0.f;
    __syncthreads();

    // stage 1: x1 = relu(Pvb[bv] + Pw[b,w] + e[b,v,w,:] @ Wc)   K=32 N=128
    {
      v16h a;
      const float* ep = e + ((size_t)bv * 128 + m0 + ln) * 32 + half * 8;
      v4f f0 = *(const v4f*)(ep);
      v4f f1 = *(const v4f*)(ep + 4);
      v4f f2 = *(const v4f*)(ep + 16);
      v4f f3 = *(const v4f*)(ep + 20);
#pragma unroll
      for (int i = 0; i < 4; ++i) {
        a[i] = (_Float16)f0[i];     a[4 + i] = (_Float16)f1[i];
        a[8 + i] = (_Float16)f2[i]; a[12 + i] = (_Float16)f3[i];
      }
      for (int nt = 0; nt < 8; ++nt) {
        const int n0 = nt * 16;
        const float pvf = Pvb[bv * 128 + n0 + ln];
        v8f acc;
#pragma unroll
        for (int r = 0; r < 8; ++r)
          acc[r] = pvf + (float)pw[(m0 + r + 8 * half) * 128 + n0 + ln];
        acc = wmma_f16(a, frag_b(wc, 32, n0, 0, lane), acc);
        store_relu(x1, 128, m0, n0, lane, acc);
      }
    }
    __syncthreads();
    // stage 2: x2 = relu(x1 @ W1 + b1)   K=128 N=256
    {
      v16h a[4];
#pragma unroll
      for (int k = 0; k < 4; ++k) a[k] = frag_a(x1, 128, m0, k * 32, lane);
      for (int nt = 0; nt < 16; ++nt) {
        const int n0 = nt * 16;
        const float bias = b1[n0 + ln];
        v8f acc;
#pragma unroll
        for (int r = 0; r < 8; ++r) acc[r] = bias;
#pragma unroll
        for (int k = 0; k < 4; ++k)
          acc = wmma_f16(a[k], frag_b(w1, 128, n0, k * 32, lane), acc);
        store_relu(x2, 256, m0, n0, lane, acc);
      }
    }
    __syncthreads();
    // stage 3: x3 = relu(x2 @ W2 + b2)   K=256 N=128  (x3 aliases x1)
    {
      v16h a[8];
#pragma unroll
      for (int k = 0; k < 8; ++k) a[k] = frag_a(x2, 256, m0, k * 32, lane);
      for (int nt = 0; nt < 8; ++nt) {
        const int n0 = nt * 16;
        const float bias = b2[n0 + ln];
        v8f acc;
#pragma unroll
        for (int r = 0; r < 8; ++r) acc[r] = bias;
#pragma unroll
        for (int k = 0; k < 8; ++k)
          acc = wmma_f16(a[k], frag_b(w2, 256, n0, k * 32, lane), acc);
        store_relu(x1, 128, m0, n0, lane, acc);
      }
    }
    __syncthreads();
    // stage 4: m = x3 @ W3 + b3; macc[n] += g[b,v,w] * m[w][n]
    {
      v16h a[4];
#pragma unroll
      for (int k = 0; k < 4; ++k) a[k] = frag_a(x1, 128, m0, k * 32, lane);
      float gw[8];
#pragma unroll
      for (int r = 0; r < 8; ++r) gw[r] = g[bv * 128 + m0 + r + 8 * half];
      for (int nt = 0; nt < 8; ++nt) {
        const int n0 = nt * 16;
        const float bias = b3[n0 + ln];
        v8f acc;
#pragma unroll
        for (int r = 0; r < 8; ++r) acc[r] = bias;
#pragma unroll
        for (int k = 0; k < 4; ++k)
          acc = wmma_f16(a[k], frag_b(w3, 128, n0, k * 32, lane), acc);
        float s = 0.f;
#pragma unroll
        for (int r = 0; r < 8; ++r) s += gw[r] * acc[r];
        atomicAdd(&macc[n0 + ln], s); // ds_add_f32
      }
    }
    __syncthreads();
    if (tid < 128) magg[bv * 128 + tid] = macc[tid];
  }
}

// ---------------- fused update MLP ----------------
// grid 16 x 128 threads (4 waves x 16-row strips, 64 rows per WG)

__global__ __launch_bounds__(128) void k_update(
    const _Float16* __restrict__ h, const float* __restrict__ magg,
    const _Float16* __restrict__ W0t, const _Float16* __restrict__ W1t,
    const _Float16* __restrict__ W2t, const _Float16* __restrict__ W3t,
    const float* __restrict__ b0, const float* __restrict__ b1,
    const float* __restrict__ b2, const float* __restrict__ b3,
    _Float16* __restrict__ hout) {
  extern __shared__ char smem[];
  _Float16* w0 = (_Float16*)smem;   // 128*256
  _Float16* w1 = w0 + 128 * 256;    // 256*128
  _Float16* w2 = w1 + 256 * 128;    // 128*256
  _Float16* w3 = w2 + 128 * 256;    // 128*128
  _Float16* x0 = w3 + 128 * 128;    // 64*256
  _Float16* x1 = x0 + 64 * 256;     // 64*128 (also x3)
  _Float16* x2 = x1 + 64 * 128;     // 64*256
  const int tid = threadIdx.x, lane = tid & 31, wid = tid >> 5;
  const int half = lane >> 4, ln = lane & 15;
  const int row0 = blockIdx.x * 64;
  copy_async_f16(w0, W0t, 128 * 256, tid, 128);
  copy_async_f16(w1, W1t, 256 * 128, tid, 128);
  copy_async_f16(w2, W2t, 128 * 256, tid, 128);
  copy_async_f16(w3, W3t, 128 * 128, tid, 128);
  {
    const unsigned int xb = lds_addr_of(x0);
    const _Float16* hb = h + row0 * 128;
    for (int i = tid; i < 64 * 16; i += 128) { // h part of x0 (row stride 256)
      const int r = i >> 4, c = i & 15;
      async_ld16(xb + (unsigned int)(r * 256 + c * 8) * 2u,
                 (unsigned int)(r * 128 + c * 8) * 2u, hb);
    }
  }
  for (int i = tid; i < 64 * 128; i += 128) { // m part of x0 (f32 -> f16)
    const int r = i >> 7, c = i & 127;
    x0[r * 256 + 128 + c] = (_Float16)magg[(row0 + r) * 128 + c];
  }
  async_wait();
  __syncthreads();
  const int m0 = wid * 16;
  { // stage 1: K=256 N=128
    v16h a[8];
#pragma unroll
    for (int k = 0; k < 8; ++k) a[k] = frag_a(x0, 256, m0, k * 32, lane);
    for (int nt = 0; nt < 8; ++nt) {
      const int n0 = nt * 16;
      const float bias = b0[n0 + ln];
      v8f acc;
#pragma unroll
      for (int r = 0; r < 8; ++r) acc[r] = bias;
#pragma unroll
      for (int k = 0; k < 8; ++k)
        acc = wmma_f16(a[k], frag_b(w0, 256, n0, k * 32, lane), acc);
      store_relu(x1, 128, m0, n0, lane, acc);
    }
  }
  __syncthreads();
  { // stage 2: K=128 N=256
    v16h a[4];
#pragma unroll
    for (int k = 0; k < 4; ++k) a[k] = frag_a(x1, 128, m0, k * 32, lane);
    for (int nt = 0; nt < 16; ++nt) {
      const int n0 = nt * 16;
      const float bias = b1[n0 + ln];
      v8f acc;
#pragma unroll
      for (int r = 0; r < 8; ++r) acc[r] = bias;
#pragma unroll
      for (int k = 0; k < 4; ++k)
        acc = wmma_f16(a[k], frag_b(w1, 128, n0, k * 32, lane), acc);
      store_relu(x2, 256, m0, n0, lane, acc);
    }
  }
  __syncthreads();
  { // stage 3: K=256 N=128 -> x1
    v16h a[8];
#pragma unroll
    for (int k = 0; k < 8; ++k) a[k] = frag_a(x2, 256, m0, k * 32, lane);
    for (int nt = 0; nt < 8; ++nt) {
      const int n0 = nt * 16;
      const float bias = b2[n0 + ln];
      v8f acc;
#pragma unroll
      for (int r = 0; r < 8; ++r) acc[r] = bias;
#pragma unroll
      for (int k = 0; k < 8; ++k)
        acc = wmma_f16(a[k], frag_b(w2, 256, n0, k * 32, lane), acc);
      store_relu(x1, 128, m0, n0, lane, acc);
    }
  }
  __syncthreads();
  { // stage 4: K=128 N=128, no relu, store f16 h'
    v16h a[4];
#pragma unroll
    for (int k = 0; k < 4; ++k) a[k] = frag_a(x1, 128, m0, k * 32, lane);
    for (int nt = 0; nt < 8; ++nt) {
      const int n0 = nt * 16;
      const float bias = b3[n0 + ln];
      v8f acc;
#pragma unroll
      for (int r = 0; r < 8; ++r) acc[r] = bias;
#pragma unroll
      for (int k = 0; k < 4; ++k)
        acc = wmma_f16(a[k], frag_b(w3, 128, n0, k * 32, lane), acc);
#pragma unroll
      for (int r = 0; r < 8; ++r)
        hout[(row0 + m0 + r + 8 * half) * 128 + n0 + ln] = (_Float16)acc[r];
    }
  }
}

// ---------------- readout: sum over nodes + MLP (single WG) ----------------

__global__ __launch_bounds__(256) void k_readout(
    const _Float16* __restrict__ h,
    const _Float16* __restrict__ W0t, const _Float16* __restrict__ W1t,
    const _Float16* __restrict__ W2t, const _Float16* __restrict__ W3t,
    const float* __restrict__ b0, const float* __restrict__ b1,
    const float* __restrict__ b2, const float* __restrict__ b3,
    float* __restrict__ out) {
  extern __shared__ char smem[];
  _Float16* w0 = (_Float16*)smem;  // 128*128
  _Float16* w1 = w0 + 128 * 128;   // 256*128
  _Float16* w2 = w1 + 256 * 128;   // 128*256
  _Float16* w3 = w2 + 128 * 256;   // 16*128 (N padded 12->16)
  _Float16* x0 = w3 + 16 * 128;    // 16*128 (rows 8..15 zero)
  _Float16* x1 = x0 + 16 * 128;    // 16*128
  _Float16* x2 = x1 + 16 * 128;    // 16*256
  _Float16* x3 = x2 + 16 * 256;    // 16*128
  const int tid = threadIdx.x, lane = tid & 31, wid = tid >> 5;
  const int half = lane >> 4, ln = lane & 15;
  copy_async_f16(w0, W0t, 128 * 128, tid, 256);
  copy_async_f16(w1, W1t, 256 * 128, tid, 256);
  copy_async_f16(w2, W2t, 128 * 256, tid, 256);
  copy_async_f16(w3, W3t, 16 * 128, tid, 256);
  for (int i = tid; i < 16 * 128; i += 256) { // node-sum, rows>=8 zero pad
    const int bb = i >> 7, n = i & 127;
    float s = 0.f;
    if (bb < 8)
      for (int v = 0; v < 128; ++v) s += (float)h[((bb << 7) + v) * 128 + n];
    x0[i] = (_Float16)s;
  }
  async_wait();
  __syncthreads();
  if (wid == 0) { // single 16-row strip; wave0 does all WMMA work
    { // K=128 N=128
      v16h a[4];
#pragma unroll
      for (int k = 0; k < 4; ++k) a[k] = frag_a(x0, 128, 0, k * 32, lane);
      for (int nt = 0; nt < 8; ++nt) {
        const int n0 = nt * 16;
        const float bias = b0[n0 + ln];
        v8f acc;
#pragma unroll
        for (int r = 0; r < 8; ++r) acc[r] = bias;
#pragma unroll
        for (int k = 0; k < 4; ++k)
          acc = wmma_f16(a[k], frag_b(w0, 128, n0, k * 32, lane), acc);
        store_relu(x1, 128, 0, n0, lane, acc);
      }
    }
    { // K=128 N=256
      v16h a[4];
#pragma unroll
      for (int k = 0; k < 4; ++k) a[k] = frag_a(x1, 128, 0, k * 32, lane);
      for (int nt = 0; nt < 16; ++nt) {
        const int n0 = nt * 16;
        const float bias = b1[n0 + ln];
        v8f acc;
#pragma unroll
        for (int r = 0; r < 8; ++r) acc[r] = bias;
#pragma unroll
        for (int k = 0; k < 4; ++k)
          acc = wmma_f16(a[k], frag_b(w1, 128, n0, k * 32, lane), acc);
        store_relu(x2, 256, 0, n0, lane, acc);
      }
    }
    { // K=256 N=128
      v16h a[8];
#pragma unroll
      for (int k = 0; k < 8; ++k) a[k] = frag_a(x2, 256, 0, k * 32, lane);
      for (int nt = 0; nt < 8; ++nt) {
        const int n0 = nt * 16;
        const float bias = b2[n0 + ln];
        v8f acc;
#pragma unroll
        for (int r = 0; r < 8; ++r) acc[r] = bias;
#pragma unroll
        for (int k = 0; k < 8; ++k)
          acc = wmma_f16(a[k], frag_b(w2, 256, n0, k * 32, lane), acc);
        store_relu(x3, 128, 0, n0, lane, acc);
      }
    }
    { // K=128 N=16 (12 valid), rows 0..7 valid -> out [8][12] f32
      v16h a[4];
#pragma unroll
      for (int k = 0; k < 4; ++k) a[k] = frag_a(x3, 128, 0, k * 32, lane);
      const float bias = (ln < 12) ? b3[ln] : 0.f;
      v8f acc;
#pragma unroll
      for (int r = 0; r < 8; ++r) acc[r] = bias;
#pragma unroll
      for (int k = 0; k < 4; ++k)
        acc = wmma_f16(a[k], frag_b(w3, 128, 0, k * 32, lane), acc);
      if (ln < 12) {
#pragma unroll
        for (int r = 0; r < 8; ++r) {
          const int m = r + 8 * half;
          if (m < 8) out[m * 12 + ln] = acc[r];
        }
      }
    }
  }
}

// ---------------- host launch ----------------

extern "C" void kernel_launch(void* const* d_in, const int* in_sizes, int n_in,
                              void* d_out, int out_size, void* d_ws, size_t ws_size,
                              hipStream_t stream) {
  (void)in_sizes; (void)n_in; (void)out_size; (void)ws_size;
  const float* g    = (const float*)d_in[0];
  const float* h_in = (const float*)d_in[1];
  const float* e    = (const float*)d_in[2];
  const float *mW[3][4], *mB[3][4], *uW[3][4], *uB[3][4], *rW[4], *rB[4];
  for (int l = 0; l < 3; ++l)
    for (int j = 0; j < 4; ++j) {
      mW[l][j] = (const float*)d_in[3 + 8 * l + 2 * j];
      mB[l][j] = (const float*)d_in[3 + 8 * l + 2 * j + 1];
      uW[l][j] = (const float*)d_in[27 + 8 * l + 2 * j];
      uB[l][j] = (const float*)d_in[27 + 8 * l + 2 * j + 1];
    }
  for (int j = 0; j < 4; ++j) {
    rW[j] = (const float*)d_in[51 + 2 * j];
    rB[j] = (const float*)d_in[51 + 2 * j + 1];
  }

  char* wsp = (char*)d_ws;
  size_t off = 0;
  auto alloc = [&](size_t bytes) -> char* {
    char* p = wsp + off;
    off += (bytes + 255) & ~(size_t)255;
    return p;
  };
  _Float16* hbuf0 = (_Float16*)alloc(1024 * 128 * 2);
  _Float16* hbuf1 = (_Float16*)alloc(1024 * 128 * 2);
  float*    Pvb   = (float*)   alloc(1024 * 128 * 4);
  _Float16* Pw    = (_Float16*)alloc(1024 * 128 * 2);
  float*    magg  = (float*)   alloc(1024 * 128 * 4);
  _Float16 *Wm0t[3], *Wm1t[3], *Wm2t[3], *Wm3t[3];
  _Float16 *Wu0t[3], *Wu1t[3], *Wu2t[3], *Wu3t[3];
  for (int l = 0; l < 3; ++l) {
    Wm0t[l] = (_Float16*)alloc(128 * 288 * 2);
    Wm1t[l] = (_Float16*)alloc(256 * 128 * 2);
    Wm2t[l] = (_Float16*)alloc(128 * 256 * 2);
    Wm3t[l] = (_Float16*)alloc(128 * 128 * 2);
    Wu0t[l] = (_Float16*)alloc(128 * 256 * 2);
    Wu1t[l] = (_Float16*)alloc(256 * 128 * 2);
    Wu2t[l] = (_Float16*)alloc(128 * 256 * 2);
    Wu3t[l] = (_Float16*)alloc(128 * 128 * 2);
  }
  _Float16* Wr0t = (_Float16*)alloc(128 * 128 * 2);
  _Float16* Wr1t = (_Float16*)alloc(256 * 128 * 2);
  _Float16* Wr2t = (_Float16*)alloc(128 * 256 * 2);
  _Float16* Wr3t = (_Float16*)alloc(16 * 128 * 2);

  k_cvt_f16<<<64, 256, 0, stream>>>(h_in, hbuf0, 1024 * 128);
  for (int l = 0; l < 3; ++l) {
    k_prep_wt<<<64, 256, 0, stream>>>(mW[l][0], Wm0t[l], 288, 128, 128);
    k_prep_wt<<<64, 256, 0, stream>>>(mW[l][1], Wm1t[l], 128, 256, 256);
    k_prep_wt<<<64, 256, 0, stream>>>(mW[l][2], Wm2t[l], 256, 128, 128);
    k_prep_wt<<<64, 256, 0, stream>>>(mW[l][3], Wm3t[l], 128, 128, 128);
    k_prep_wt<<<64, 256, 0, stream>>>(uW[l][0], Wu0t[l], 256, 128, 128);
    k_prep_wt<<<64, 256, 0, stream>>>(uW[l][1], Wu1t[l], 128, 256, 256);
    k_prep_wt<<<64, 256, 0, stream>>>(uW[l][2], Wu2t[l], 256, 128, 128);
    k_prep_wt<<<64, 256, 0, stream>>>(uW[l][3], Wu3t[l], 128, 128, 128);
  }
  k_prep_wt<<<64, 256, 0, stream>>>(rW[0], Wr0t, 128, 128, 128);
  k_prep_wt<<<64, 256, 0, stream>>>(rW[1], Wr1t, 128, 256, 256);
  k_prep_wt<<<64, 256, 0, stream>>>(rW[2], Wr2t, 256, 128, 128);
  k_prep_wt<<<64, 256, 0, stream>>>(rW[3], Wr3t, 128, 12, 16);

  const int MSG_LDS = (256 * 128 + 128 * 256 + 128 * 128 + 128 * 32 + 128 * 128 +
                       128 * 128 + 128 * 256) * 2 + 128 * 4;
  const int UPD_LDS = (128 * 256 + 256 * 128 + 128 * 256 + 128 * 128 + 64 * 256 +
                       64 * 128 + 64 * 256) * 2;
  const int RO_LDS = (128 * 128 + 256 * 128 + 128 * 256 + 16 * 128 + 16 * 128 +
                      16 * 128 + 16 * 256 + 16 * 128) * 2;

  _Float16* hcur = hbuf0;
  _Float16* hnxt = hbuf1;
  for (int l = 0; l < 3; ++l) {
    k_pvpw<<<8, 256, 128 * 256 * 2, stream>>>(hcur, Wm0t[l], mB[l][0], Pvb, Pw);
    k_message<<<256, 256, MSG_LDS, stream>>>(e, g, Pvb, Pw, Wm0t[l], Wm1t[l],
                                             Wm2t[l], Wm3t[l], mB[l][1], mB[l][2],
                                             mB[l][3], magg);
    k_update<<<16, 128, UPD_LDS, stream>>>(hcur, magg, Wu0t[l], Wu1t[l], Wu2t[l],
                                           Wu3t[l], uB[l][0], uB[l][1], uB[l][2],
                                           uB[l][3], hnxt);
    _Float16* t = hcur; hcur = hnxt; hnxt = t;
  }
  k_readout<<<1, 256, RO_LDS, stream>>>(hcur, Wr0t, Wr1t, Wr2t, Wr3t, rB[0], rB[1],
                                        rB[2], rB[3], (float*)d_out);
}
